// ConvGRUSeparator2d_85950885528365
// MI455X (gfx1250) — matrix operand order repaired and checked by reading, not compiled, not run on hardware
//
#include <hip/hip_runtime.h>

// Problem constants (match reference)
#define BB 4
#define CC 128
#define TT 200
#define FF 256
#define LL 4
#define KK 3
#define EPSF 1e-6f
#define NTHREADS 256   // 8 wave32 waves; wave w owns channel rows [16w, 16w+16)
#define NWAVES 8

typedef _Float16 v16h __attribute__((ext_vector_type(16)));
typedef _Float16 v8h  __attribute__((ext_vector_type(8)));
typedef float    v8f  __attribute__((ext_vector_type(8)));

// Dynamic LDS layout (bytes):
//   hS   [C][F] f32   : 131072   (carry state = previous cell output)
//   xnT  [F][C] f16   :  65536   (normalized x_t, transposed for B-fragments)
//   hmT  [F][C] f16   :  65536   (normalized+band-mixed h, transposed)
//   msh  [F] f32      :   1024   (h RMS stats; neighbors needed by the conv)
//   wIn/wHid/wOut [C] :   1536
//   mixW [C*K]        :   1536
//   mixb [C]          :    512
//   xb/hb [3C] each   :   3072
constexpr size_t SMEM_BYTES =
    (size_t)CC * FF * 4 + (size_t)FF * CC * 2 * 2 + FF * 4 +
    3 * CC * 4 + CC * KK * 4 + CC * 4 + 2 * 3 * CC * 4;   // = 269824

// ---- WMMA fragment loaders (wave32, 16-bit layouts per CDNA5 ISA 7.12.2) ----
// A 16x32 (MxK): lane row m=lane&15; lanes 0-15 hold K=k0..k0+7 & k0+16..k0+23,
// lanes 16-31 hold K=k0+8..k0+15 & k0+24..k0+31.
__device__ __forceinline__ v16h load_frag_a(const _Float16* __restrict__ w,
                                            int m, int k0, int lane) {
  const int ka = k0 + ((lane & 16) ? 8 : 0);
  const _Float16* p = w + (size_t)m * CC + ka;
  v8h lo = *(const v8h*)(p);
  v8h hi = *(const v8h*)(p + 16);
  v16h a;
#pragma unroll
  for (int i = 0; i < 8; ++i) { a[i] = lo[i]; a[i + 8] = hi[i]; }
  return a;
}

// B 32x16 (KxN) from LDS table tbl[F][C] (K contiguous): column n=lane&15,
// lanes 0-15 hold K=k0..k0+7 & k0+16..k0+23, lanes 16-31 hold K=k0+8..k0+15 & k0+24..k0+31.
__device__ __forceinline__ v16h load_frag_b(const _Float16* tbl,
                                            int f, int k0, int lane) {
  const int kb = k0 + ((lane & 16) ? 8 : 0);
  const _Float16* p = tbl + f * CC + kb;
  v8h lo = *(const v8h*)(p);
  v8h hi = *(const v8h*)(p + 16);
  v16h b;
#pragma unroll
  for (int i = 0; i < 8; ++i) { b[i] = lo[i]; b[i + 8] = hi[i]; }
  return b;
}

__device__ __forceinline__ float sigmoidf_(float v) {
  return 1.0f / (1.0f + __expf(-v));
}

// ---- Pre-kernel: cache xW/hW as f16 in workspace (A-operands for WMMA) ----
__global__ void convert_weights_kernel(const float* __restrict__ xW,
                                       const float* __restrict__ hW,
                                       _Float16* __restrict__ w16, int S) {
  int i = blockIdx.x * blockDim.x + threadIdx.x;
  int total = LL * S;
  if (i < total) {
    int l = i / S, r = i - l * S;
    w16[(size_t)l * 2 * S + r]     = (_Float16)xW[i];
    w16[(size_t)l * 2 * S + S + r] = (_Float16)hW[i];
  }
}

// ---- Persistent per-batch ConvGRU layer kernel ----
__global__ void __launch_bounds__(NTHREADS) gru_layer_kernel(
    const float* __restrict__ x, float* __restrict__ y,
    const _Float16* __restrict__ wx16, const _Float16* __restrict__ wh16,
    const float* __restrict__ in_w, const float* __restrict__ hid_w,
    const float* __restrict__ out_w, const float* __restrict__ xb,
    const float* __restrict__ mixW, const float* __restrict__ mixb,
    const float* __restrict__ hb) {
  extern __shared__ unsigned char smem_raw[];
  float*    hS  = (float*)smem_raw;                    // [C][F]
  _Float16* xnT = (_Float16*)(smem_raw + CC * FF * 4); // [F][C]
  _Float16* hmT = xnT + FF * CC;                       // [F][C]
  float* msh = (float*)(hmT + FF * CC);                // [F]
  float* wIn = msh + FF;
  float* wHid = wIn + CC;
  float* wOut = wHid + CC;
  float* mWs = wOut + CC;        // [C*K]
  float* mBs = mWs + CC * KK;    // [C]
  float* xBs = mBs + CC;         // [3C]
  float* hBs = xBs + 3 * CC;     // [3C]

  const int tid  = threadIdx.x;
  const int lane = tid & 31;
  const int wave = tid >> 5;     // 0..7
  const int b    = blockIdx.x;

  for (int i = tid; i < CC; i += NTHREADS) {
    wIn[i] = in_w[i]; wHid[i] = hid_w[i]; wOut[i] = out_w[i]; mBs[i] = mixb[i];
  }
  for (int i = tid; i < 3 * CC; i += NTHREADS) { xBs[i] = xb[i]; hBs[i] = hb[i]; }
  for (int i = tid; i < CC * KK; i += NTHREADS) mWs[i] = mixW[i];
  for (int i = tid; i < CC * FF; i += NTHREADS) hS[i] = 0.0f;

  // Hoist this wave's A-fragments (weight rows c0..c0+15, all gates, all K)
  // into registers for the whole sequence: 24 x v16h = 192 VGPRs.
  const int c0 = wave * 16;
  v16h aX[12], aH[12];
  {
    const int mrow = c0 + (lane & 15);
#pragma unroll
    for (int g = 0; g < 3; ++g) {
#pragma unroll
      for (int ks = 0; ks < 4; ++ks) {
        aX[g * 4 + ks] = load_frag_a(wx16, g * CC + mrow, ks * 32, lane);
        aH[g * 4 + ks] = load_frag_a(wh16, g * CC + mrow, ks * 32, lane);
      }
    }
  }
  __syncthreads();

  const size_t strideC = (size_t)TT * FF;  // channel stride in x/y
  const float* xb_base = x + (size_t)b * CC * strideC;
  float*       yb_base = y + (size_t)b * CC * strideC;

  const int fcol = tid;  // one thread per frequency column (F == NTHREADS)

  for (int t = 0; t < TT; ++t) {
    const float* xt = xb_base + (size_t)t * FF;
    float*       yt = yb_base + (size_t)t * FF;

    // Prefetch next timestep's x slab (one channel row per pair of threads).
    if (t + 1 < TT)
      __builtin_prefetch(xt + strideC * (size_t)(tid & (CC - 1)) + FF, 0, 0);

    // Pass A: own-column stats. sx stays in a register; msh published to LDS
    // (pass B reads neighbor columns for the depthwise conv).
    float sx = 0.0f;
    {
      float sh = 0.0f;
#pragma unroll 4
      for (int c = 0; c < CC; ++c) {
        float xv = xt[(size_t)c * strideC + fcol];
        float hv = hS[c * FF + fcol];
        sx += xv * xv;
        sh += hv * hv;
      }
      msh[fcol] = sh;
    }
    __syncthreads();

    // Pass B: build xnT = rms(x)*in_w and hmT = dwconv(rms(h)*hid_w)+mixb.
    {
      const float rx  = rsqrtf(sx * (1.0f / CC) + EPSF);
      const float rh0 = (fcol > 0)      ? rsqrtf(msh[fcol - 1] * (1.0f / CC) + EPSF) : 0.0f;
      const float rh1 =                   rsqrtf(msh[fcol]     * (1.0f / CC) + EPSF);
      const float rh2 = (fcol < FF - 1) ? rsqrtf(msh[fcol + 1] * (1.0f / CC) + EPSF) : 0.0f;
#pragma unroll 2
      for (int c = 0; c < CC; ++c) {
        float xv = xt[(size_t)c * strideC + fcol];
        xnT[fcol * CC + c] = (_Float16)(xv * rx * wIn[c]);
        const float wh = wHid[c];
        float hm = mBs[c];
        if (fcol > 0)      hm += mWs[c * KK + 0] * hS[c * FF + fcol - 1] * rh0 * wh;
                           hm += mWs[c * KK + 1] * hS[c * FF + fcol    ] * rh1 * wh;
        if (fcol < FF - 1) hm += mWs[c * KK + 2] * hS[c * FF + fcol + 1] * rh2 * wh;
        hmT[fcol * CC + c] = (_Float16)hm;
      }
    }
    __syncthreads();

    // Pass D: WMMA gate GEMMs. Wave sweeps its channel slice across all 16
    // f-tiles; A-operands are register-resident, B-operands stream from LDS.
    for (int fi = 0; fi < 16; ++fi) {
      const int f0 = fi * 16;
      v8f axr = {}, axz = {}, axc = {}, ahr = {}, ahz = {}, ahc = {};
#pragma unroll
      for (int ks = 0; ks < 4; ++ks) {
        const int k0 = ks * 32;
        const v16h bx = load_frag_b(xnT, f0 + (lane & 15), k0, lane);
        const v16h bh = load_frag_b(hmT, f0 + (lane & 15), k0, lane);
        axr = __builtin_amdgcn_wmma_f32_16x16x32_f16(false, aX[0 + ks], false, bx, (short)0, axr, false, false);
        axz = __builtin_amdgcn_wmma_f32_16x16x32_f16(false, aX[4 + ks], false, bx, (short)0, axz, false, false);
        axc = __builtin_amdgcn_wmma_f32_16x16x32_f16(false, aX[8 + ks], false, bx, (short)0, axc, false, false);
        ahr = __builtin_amdgcn_wmma_f32_16x16x32_f16(false, aH[0 + ks], false, bh, (short)0, ahr, false, false);
        ahz = __builtin_amdgcn_wmma_f32_16x16x32_f16(false, aH[4 + ks], false, bh, (short)0, ahz, false, false);
        ahc = __builtin_amdgcn_wmma_f32_16x16x32_f16(false, aH[8 + ks], false, bh, (short)0, ahc, false, false);
      }
      // C/D layout: VGPR v holds M=v (lanes 0-15) / M=v+8 (lanes 16-31).
      const int f    = f0 + (lane & 15);
      const int mofs = (lane & 16) ? 8 : 0;
#pragma unroll
      for (int v = 0; v < 8; ++v) {
        const int c = c0 + v + mofs;
        const float r = sigmoidf_((axr[v] + xBs[c]) + (ahr[v] + hBs[c]));
        const float z = sigmoidf_((axz[v] + xBs[CC + c]) + (ahz[v] + hBs[CC + c]));
        const float cand = tanhf((axc[v] + xBs[2 * CC + c]) + r * (ahc[v] + hBs[2 * CC + c]));
        const float hold = hS[c * FF + f];
        hS[c * FF + f] = (1.0f - z) * cand + z * hold;  // h_new (pre-residual)
      }
    }
    __syncthreads();

    // Pass E: out-RMS of (h_new + x_t), write y_t, carry h <- out.
    // Purely thread-local (own column), so no extra barrier needed.
    {
      float so = 0.0f;
#pragma unroll 4
      for (int c = 0; c < CC; ++c) {
        float u = hS[c * FF + fcol] + xt[(size_t)c * strideC + fcol];
        so += u * u;
      }
      const float ro = rsqrtf(so * (1.0f / CC) + EPSF);
#pragma unroll 2
      for (int c = 0; c < CC; ++c) {
        float u = hS[c * FF + fcol] + xt[(size_t)c * strideC + fcol];
        float o = u * ro * wOut[c];
        yt[(size_t)c * strideC + fcol] = o;
        hS[c * FF + fcol] = o;
      }
    }
    // Next step's pass A reads only the thread's own column; the barrier
    // after pass A orders these writes before neighbor reads in pass B.
  }
}

extern "C" void kernel_launch(void* const* d_in, const int* in_sizes, int n_in,
                              void* d_out, int out_size, void* d_ws, size_t ws_size,
                              hipStream_t stream) {
  (void)in_sizes; (void)n_in; (void)out_size; (void)ws_size;
  const float* x     = (const float*)d_in[0];
  const float* in_w  = (const float*)d_in[1];
  const float* hid_w = (const float*)d_in[2];
  const float* out_w = (const float*)d_in[3];
  const float* xW    = (const float*)d_in[4];
  const float* xb    = (const float*)d_in[5];
  const float* mixW  = (const float*)d_in[6];
  const float* mixb  = (const float*)d_in[7];
  const float* hW    = (const float*)d_in[8];
  const float* hb    = (const float*)d_in[9];
  float* out = (float*)d_out;

  const int S = 3 * CC * CC;  // 49152 elems per weight matrix per layer
  _Float16* w16 = (_Float16*)d_ws;                          // LL*2*S f16
  float* ybuf = (float*)((char*)d_ws + (size_t)LL * 2 * S * sizeof(_Float16));

  (void)hipFuncSetAttribute((const void*)gru_layer_kernel,
                            hipFuncAttributeMaxDynamicSharedMemorySize,
                            (int)SMEM_BYTES);

  const int total = LL * S;
  convert_weights_kernel<<<(total + 255) / 256, 256, 0, stream>>>(xW, hW, w16, S);

  struct PP { const float* src; float* dst; };
  const PP pp[LL] = { {x, ybuf}, {ybuf, out}, {out, ybuf}, {ybuf, out} };

  for (int l = 0; l < LL; ++l) {
    gru_layer_kernel<<<dim3(BB), dim3(NTHREADS), SMEM_BYTES, stream>>>(
        pp[l].src, pp[l].dst,
        w16 + (size_t)l * 2 * S, w16 + (size_t)l * 2 * S + S,
        in_w + (size_t)l * CC, hid_w + (size_t)l * CC, out_w + (size_t)l * CC,
        xb + (size_t)l * 3 * CC, mixW + (size_t)l * CC * KK,
        mixb + (size_t)l * CC, hb + (size_t)l * 3 * CC);
  }
}
